// CausalSelfAttention_30880814858932
// MI455X (gfx1250) — matrix operand order repaired and checked by reading, not compiled, not run on hardware
//
#include <hip/hip_runtime.h>

#define BATCH 2
#define TSEQ 2048
#define CDIM 1024
#define NHEAD 16
#define HDIM 64
#define C3 (3 * CDIM)
#define MROWS (BATCH * TSEQ)

typedef __attribute__((ext_vector_type(16))) __bf16 v16bf;
typedef __attribute__((ext_vector_type(8)))  float  v8f;
typedef unsigned int v4u __attribute__((ext_vector_type(4)));
typedef int          v8i __attribute__((ext_vector_type(8)));
typedef int          v4i __attribute__((ext_vector_type(4)));

union FragA { uint4 q[2]; v16bf v; };
union Half8 { uint4 q; __bf16 h[8]; };

#if defined(__has_builtin)
#if __has_builtin(__builtin_amdgcn_tensor_load_to_lds)
#define HAVE_TDM 1
#endif
#endif
#ifndef HAVE_TDM
#define HAVE_TDM 0
#endif

__device__ __forceinline__ v8f wmma_bf16(v16bf a, v16bf b, v8f c) {
  return __builtin_amdgcn_wmma_f32_16x16x32_bf16(false, a, false, b, (short)0, c, false, false);
}

#if HAVE_TDM
// ---------------------------------------------------------------------------
// TDM: 2-D tile load (bf16 elements) Global -> LDS with 16B pad per 128B row
// (pad_interval = 32 DWORDs, pad_amount = 4 DWORDs) to produce LQ=72-half rows.
// D# layout per CDNA5 ISA ch.8: group0 = count|lds_addr|global_addr|type=2,
// group1 = data_size/pad cfg, tensor dims, tile dims, dim0 stride.
// Issue from one wave only (TDM ignores EXEC); track with TENSORcnt.
// ---------------------------------------------------------------------------
__device__ __forceinline__ void tdm_load_2d_bf16(unsigned int lds_off,
                                                 const void* gptr,
                                                 unsigned int width,   // elems, must be 64
                                                 unsigned int rows,
                                                 unsigned int stride)  // elems
{
  unsigned long long ga = (unsigned long long)gptr;
  v4u g0;
  g0.x = 1u;                                   // count=1 (valid user D#)
  g0.y = lds_off;                              // lds_addr (bytes)
  g0.z = (unsigned int)ga;                     // global_addr[31:0]
  g0.w = (unsigned int)(ga >> 32) | 0x80000000u;  // global_addr msbs | type=2
  v8i g1;
  g1[0] = (int)((1u << 16) | (1u << 20) | (4u << 22) | (3u << 25));
  //            data_size=2B  pad_en      every 32 dw   pad 4 dw
  g1[1] = (int)((width & 0xFFFFu) << 16);                          // tensor_dim0 lo
  g1[2] = (int)((width >> 16) | ((rows & 0xFFFFu) << 16));         // dim0 hi | dim1 lo
  g1[3] = (int)((rows >> 16) | (width << 16));                     // dim1 hi | tile_dim0
  g1[4] = (int)(rows & 0xFFFFu);                                   // tile_dim1 | tile_dim2=0
  g1[5] = (int)stride;                                             // dim0_stride lo32
  g1[6] = 0;
  g1[7] = 0;
  v4i z4 = {0, 0, 0, 0};
#if defined(__clang_major__) && (__clang_major__ >= 23)
  v8i z8 = {0, 0, 0, 0, 0, 0, 0, 0};
  __builtin_amdgcn_tensor_load_to_lds(g0, g1, z4, z4, z8, 0);
#else
  __builtin_amdgcn_tensor_load_to_lds(g0, g1, z4, z4, 0);
#endif
}
#endif  // HAVE_TDM

// ---------------------------------------------------------------------------
// fp32 -> bf16 elementwise convert (vectorized 4/thread)
// ---------------------------------------------------------------------------
__global__ __launch_bounds__(256) void cvt_f32_bf16(const float* __restrict__ in,
                                                    __bf16* __restrict__ out) {
  size_t i = ((size_t)blockIdx.x * 256 + threadIdx.x) * 4;
  float4 f = *reinterpret_cast<const float4*>(in + i);
  out[i + 0] = (__bf16)f.x;
  out[i + 1] = (__bf16)f.y;
  out[i + 2] = (__bf16)f.z;
  out[i + 3] = (__bf16)f.w;
}

// ---------------------------------------------------------------------------
// C = A(bf16, MxK row-major) * B(fp32, KxN row-major, converted to bf16) + bias
// Block tile 128x128x32, 8 waves in 2(M) x 4(N) grid, wave tile 64x32.
// Double-buffered LDS: global fetch for tile k+1 overlaps WMMA on tile k.
// ---------------------------------------------------------------------------
template <bool OUT_BF16>
__global__ __launch_bounds__(256) void gemm_bf16_kernel(
    const __bf16* __restrict__ A, const float* __restrict__ Bw,
    const float* __restrict__ bias, void* __restrict__ Cout,
    int N, int K) {
  constexpr int BM = 128, BN = 128, BK = 32;
  constexpr int LDA = BK + 8;   // halves; rows stay 16B aligned (80B stride)
  constexpr int LDB = BK + 8;
  __shared__ __attribute__((aligned(16))) __bf16 As[2][BM * LDA];
  __shared__ __attribute__((aligned(16))) __bf16 Bs[2][BN * LDB];  // [n][k]

  const int tid  = threadIdx.x;
  const int lane = tid & 31;
  const int wave = tid >> 5;
  const int wm   = wave & 1;    // M offset wm*64
  const int wn   = wave >> 1;   // N offset wn*32
  const int half = lane >> 4;
  const int l16  = lane & 15;
  const int m0   = blockIdx.y * BM;
  const int n0   = blockIdx.x * BN;

  // fixed per-thread staging coordinates
  const int ar  = tid >> 2;         // A rows ar and ar+64
  const int akc = (tid & 3) * 8;    // 16B chunk within row
  const int bkr = tid >> 5;         // B k-rows bkr + j*8
  const int bnc = (tid & 31) * 4;   // 4-float chunk along n

  const __bf16* aG = A + (size_t)(m0 + ar) * K + akc;
  const float*  bG = Bw + (size_t)bkr * N + n0 + bnc;

  const v8f vzero = {0.f, 0.f, 0.f, 0.f, 0.f, 0.f, 0.f, 0.f};
  v8f acc[4][2];
#pragma unroll
  for (int mt = 0; mt < 4; ++mt)
#pragma unroll
    for (int nt = 0; nt < 2; ++nt) acc[mt][nt] = vzero;

  uint4  aR0, aR1;
  float4 bR[4];
  aR0 = *reinterpret_cast<const uint4*>(aG);
  aR1 = *reinterpret_cast<const uint4*>(aG + (size_t)64 * K);
#pragma unroll
  for (int j = 0; j < 4; ++j)
    bR[j] = *reinterpret_cast<const float4*>(bG + (size_t)(j * 8) * N);

  auto stage = [&](int buf) {
    *reinterpret_cast<uint4*>(&As[buf][ar * LDA + akc]) = aR0;
    *reinterpret_cast<uint4*>(&As[buf][(ar + 64) * LDA + akc]) = aR1;
#pragma unroll
    for (int j = 0; j < 4; ++j) {
      Bs[buf][(bnc + 0) * LDB + bkr + j * 8] = (__bf16)bR[j].x;
      Bs[buf][(bnc + 1) * LDB + bkr + j * 8] = (__bf16)bR[j].y;
      Bs[buf][(bnc + 2) * LDB + bkr + j * 8] = (__bf16)bR[j].z;
      Bs[buf][(bnc + 3) * LDB + bkr + j * 8] = (__bf16)bR[j].w;
    }
  };
  stage(0);
  __syncthreads();

  int cur = 0;
  for (int k0 = 0; k0 < K; k0 += BK) {
    const bool nxt = (k0 + BK) < K;
    if (nxt) {  // prefetch next tile into registers while computing
      aR0 = *reinterpret_cast<const uint4*>(aG + k0 + BK);
      aR1 = *reinterpret_cast<const uint4*>(aG + (size_t)64 * K + k0 + BK);
#pragma unroll
      for (int j = 0; j < 4; ++j)
        bR[j] = *reinterpret_cast<const float4*>(bG + (size_t)(k0 + BK + j * 8) * N);
    }

    FragA af[4], bf_[2];
#pragma unroll
    for (int mt = 0; mt < 4; ++mt) {
      const __bf16* ap = &As[cur][(wm * 64 + mt * 16 + l16) * LDA + half * 8];
      af[mt].q[0] = *reinterpret_cast<const uint4*>(ap);
      af[mt].q[1] = *reinterpret_cast<const uint4*>(ap + 16);
    }
#pragma unroll
    for (int nt = 0; nt < 2; ++nt) {
      const __bf16* bp = &Bs[cur][(wn * 32 + nt * 16 + l16) * LDB + half * 8];
      bf_[nt].q[0] = *reinterpret_cast<const uint4*>(bp);
      bf_[nt].q[1] = *reinterpret_cast<const uint4*>(bp + 16);
    }
#pragma unroll
    for (int mt = 0; mt < 4; ++mt)
#pragma unroll
      for (int nt = 0; nt < 2; ++nt)
        acc[mt][nt] = wmma_bf16(af[mt].v, bf_[nt].v, acc[mt][nt]);

    if (nxt) stage(cur ^ 1);
    __syncthreads();
    cur ^= 1;
  }

#pragma unroll
  for (int mt = 0; mt < 4; ++mt)
#pragma unroll
    for (int nt = 0; nt < 2; ++nt) {
      int   n  = n0 + wn * 32 + nt * 16 + l16;
      float bv = bias[n];
#pragma unroll
      for (int r = 0; r < 8; ++r) {
        int   m = m0 + wm * 64 + mt * 16 + half * 8 + r;
        float v = acc[mt][nt][r] + bv;
        if (OUT_BF16)
          reinterpret_cast<__bf16*>(Cout)[(size_t)m * N + n] = (__bf16)v;
        else
          reinterpret_cast<float*>(Cout)[(size_t)m * N + n] = v;
      }
    }
}

// ---------------------------------------------------------------------------
// Flash attention with online softmax. Block = one (b,h) and 128 q rows,
// 8 waves x 16 q rows. Q and K tiles arrive via the Tensor Data Mover
// (wave 0 issues, TENSORcnt tracked); V staged manually (needs transpose).
// ---------------------------------------------------------------------------
__global__ __launch_bounds__(256) void attn_kernel(const __bf16* __restrict__ qkv,
                                                   __bf16* __restrict__ yb) {
  constexpr int BQ = 128, BKV = 64, LQ = HDIM + 8, LV = BKV + 8;
  __shared__ __attribute__((aligned(16))) __bf16 Qs[BQ * LQ];        // [q][d]
  __shared__ __attribute__((aligned(16))) __bf16 Ks[BKV * LQ];       // [k][d]
  __shared__ __attribute__((aligned(16))) __bf16 Vt[HDIM * LV];      // [d][k]
  __shared__ __attribute__((aligned(16))) __bf16 Ps[8 * 16 * LQ];    // per-wave P

  const int tid  = threadIdx.x;
  const int lane = tid & 31;
  const int wave = tid >> 5;
  const int half = lane >> 4;
  const int l16  = lane & 15;
  const int qb   = blockIdx.x;
  const int bidx = blockIdx.y / NHEAD;
  const int h    = blockIdx.y % NHEAD;

  const __bf16* base = qkv + (size_t)bidx * TSEQ * C3 + h * HDIM;

#if HAVE_TDM
  const unsigned int qs_off = (unsigned int)(unsigned long long)(void*)Qs;
  const unsigned int ks_off = (unsigned int)(unsigned long long)(void*)Ks;
  if (wave == 0)
    tdm_load_2d_bf16(qs_off, (const void*)(base + (size_t)(qb * BQ) * C3),
                     HDIM, BQ, C3);
#else
  for (int i = tid; i < (BQ * HDIM) / 8; i += 256) {
    int r = i >> 3, dc = (i & 7) * 8;
    *reinterpret_cast<uint4*>(&Qs[r * LQ + dc]) =
        *reinterpret_cast<const uint4*>(base + (size_t)(qb * BQ + r) * C3 + dc);
  }
#endif

  const int q0 = qb * BQ + wave * 16;
  float m_run[8], l_run[8];
  const v8f vzero = {0.f, 0.f, 0.f, 0.f, 0.f, 0.f, 0.f, 0.f};
  v8f o[4];
#pragma unroll
  for (int nt = 0; nt < 4; ++nt) o[nt] = vzero;
#pragma unroll
  for (int r = 0; r < 8; ++r) { m_run[r] = -1e30f; l_run[r] = 0.f; }

  const int nkb = 2 * (qb + 1);  // k-blocks up to causal frontier of this q-tile
  for (int kb = 0; kb < nkb; ++kb) {
    __syncthreads();  // previous iteration's compute done -> LDS reusable
#if HAVE_TDM
    if (wave == 0)
      tdm_load_2d_bf16(ks_off,
                       (const void*)(base + CDIM + (size_t)(kb * BKV) * C3),
                       HDIM, BKV, C3);
#endif
    for (int i = tid; i < (BKV * HDIM) / 8; i += 256) {
      int kk = i >> 3, dc = (i & 7) * 8;
      const __bf16* src = base + (size_t)(kb * BKV + kk) * C3 + dc;
#if !HAVE_TDM
      *reinterpret_cast<uint4*>(&Ks[kk * LQ + dc]) =
          *reinterpret_cast<const uint4*>(src + CDIM);
#endif
      Half8 vv;
      vv.q = *reinterpret_cast<const uint4*>(src + 2 * CDIM);
#pragma unroll
      for (int j = 0; j < 8; ++j) Vt[(dc + j) * LV + kk] = vv.h[j];
    }
#if HAVE_TDM
    if (wave == 0) __builtin_amdgcn_s_wait_tensorcnt((short)0);  // Q (+K) landed
#endif
    __syncthreads();
    if (kb * BKV > q0 + 15) continue;  // wave-uniform causal skip

    // S = Q * K^T  (M=16 rows of this wave, N=64, K=d=64)
    v8f s[4];
#pragma unroll
    for (int nt = 0; nt < 4; ++nt) s[nt] = vzero;
#pragma unroll
    for (int ks = 0; ks < 2; ++ks) {
      FragA a;
      const __bf16* ap = &Qs[(wave * 16 + l16) * LQ + ks * 32 + half * 8];
      a.q[0] = *reinterpret_cast<const uint4*>(ap);
      a.q[1] = *reinterpret_cast<const uint4*>(ap + 16);
#pragma unroll
      for (int nt = 0; nt < 4; ++nt) {
        FragA bm;
        const __bf16* bp = &Ks[(nt * 16 + l16) * LQ + ks * 32 + half * 8];
        bm.q[0] = *reinterpret_cast<const uint4*>(bp);
        bm.q[1] = *reinterpret_cast<const uint4*>(bp + 16);
        s[nt] = wmma_bf16(a.v, bm.v, s[nt]);
      }
    }

    // scale + causal mask + tile row-max (N striped across 16 lanes)
    float rmax[8];
#pragma unroll
    for (int r = 0; r < 8; ++r) rmax[r] = -1e30f;
#pragma unroll
    for (int nt = 0; nt < 4; ++nt) {
      int col = kb * BKV + nt * 16 + l16;
#pragma unroll
      for (int r = 0; r < 8; ++r) {
        int   qrow = q0 + half * 8 + r;
        float sv   = s[nt][r] * 0.125f;  // 1/sqrt(64)
        sv         = (col > qrow) ? -1e30f : sv;
        s[nt][r]   = sv;
        rmax[r]    = fmaxf(rmax[r], sv);
      }
    }
#pragma unroll
    for (int r = 0; r < 8; ++r) {
      rmax[r] = fmaxf(rmax[r], __shfl_xor(rmax[r], 1, 16));
      rmax[r] = fmaxf(rmax[r], __shfl_xor(rmax[r], 2, 16));
      rmax[r] = fmaxf(rmax[r], __shfl_xor(rmax[r], 4, 16));
      rmax[r] = fmaxf(rmax[r], __shfl_xor(rmax[r], 8, 16));
    }
    float rsum[8];
#pragma unroll
    for (int r = 0; r < 8; ++r) {
      float mn   = fmaxf(m_run[r], rmax[r]);
      float alph = __expf(m_run[r] - mn);
      m_run[r]   = mn;
      l_run[r]  *= alph;
      rsum[r]    = 0.f;
#pragma unroll
      for (int nt = 0; nt < 4; ++nt) o[nt][r] *= alph;
    }
    // P = exp(S - m); spill to per-wave LDS (bf16) to re-shape into A-frag layout
    __bf16* pw = &Ps[wave * 16 * LQ];
#pragma unroll
    for (int nt = 0; nt < 4; ++nt)
#pragma unroll
      for (int r = 0; r < 8; ++r) {
        float p = __expf(s[nt][r] - m_run[r]);
        rsum[r] += p;
        pw[(half * 8 + r) * LQ + nt * 16 + l16] = (__bf16)p;
      }
#pragma unroll
    for (int r = 0; r < 8; ++r) {
      rsum[r] += __shfl_xor(rsum[r], 1, 16);
      rsum[r] += __shfl_xor(rsum[r], 2, 16);
      rsum[r] += __shfl_xor(rsum[r], 4, 16);
      rsum[r] += __shfl_xor(rsum[r], 8, 16);
      l_run[r] += rsum[r];
    }
    // O += P * V  (K = kv index = 64, N = d = 64)
#pragma unroll
    for (int ks = 0; ks < 2; ++ks) {
      FragA a;
      const __bf16* ap = &pw[l16 * LQ + ks * 32 + half * 8];
      a.q[0] = *reinterpret_cast<const uint4*>(ap);
      a.q[1] = *reinterpret_cast<const uint4*>(ap + 16);
#pragma unroll
      for (int nt = 0; nt < 4; ++nt) {
        FragA bm;
        const __bf16* bp = &Vt[(nt * 16 + l16) * LV + ks * 32 + half * 8];
        bm.q[0] = *reinterpret_cast<const uint4*>(bp);
        bm.q[1] = *reinterpret_cast<const uint4*>(bp + 16);
        o[nt] = wmma_bf16(a.v, bm.v, o[nt]);
      }
    }
  }

  // normalize and store y (bf16) in [B,T,C] with head offset
#pragma unroll
  for (int r = 0; r < 8; ++r) {
    float  inv = 1.0f / l_run[r];
    int    t   = q0 + half * 8 + r;
    size_t ob  = ((size_t)bidx * TSEQ + t) * CDIM + h * HDIM;
#pragma unroll
    for (int nt = 0; nt < 4; ++nt)
      yb[ob + nt * 16 + l16] = (__bf16)(o[nt][r] * inv);
  }
}

// ---------------------------------------------------------------------------
extern "C" void kernel_launch(void* const* d_in, const int* in_sizes, int n_in,
                              void* d_out, int out_size, void* d_ws, size_t ws_size,
                              hipStream_t stream) {
  const float* x      = (const float*)d_in[0];
  const float* w_qkv  = (const float*)d_in[1];
  const float* b_qkv  = (const float*)d_in[2];
  const float* w_proj = (const float*)d_in[3];
  const float* b_proj = (const float*)d_in[4];

  // workspace: xb (8MB) | qkv bf16 (25MB) | y bf16 (8MB)
  __bf16* xb  = (__bf16*)d_ws;
  __bf16* qkv = xb + (size_t)MROWS * CDIM;
  __bf16* yb  = qkv + (size_t)MROWS * C3;

  cvt_f32_bf16<<<(MROWS * CDIM) / (256 * 4), 256, 0, stream>>>(x, xb);

  gemm_bf16_kernel<true><<<dim3(C3 / 128, MROWS / 128), 256, 0, stream>>>(
      xb, w_qkv, b_qkv, qkv, C3, CDIM);

  attn_kernel<<<dim3(TSEQ / 128, BATCH * NHEAD), 256, 0, stream>>>(qkv, yb);

  gemm_bf16_kernel<false><<<dim3(CDIM / 128, MROWS / 128), 256, 0, stream>>>(
      yb, w_proj, b_proj, d_out, CDIM, CDIM);
}
